// MultiheadAttention_56538949484963
// MI455X (gfx1250) — matrix operand order repaired and checked by reading, not compiled, not run on hardware
//
#include <hip/hip_runtime.h>

// ---------------------------------------------------------------------------
// MHA forward for MI455X (gfx1250).
//   - one-time f32->bf16 conversion of activations, and convert+transpose of
//     weights (so GEMM B-fragments are K-contiguous)
//   - bf16 WMMA GEMMs with tiles staged via GLOBAL_LOAD_ASYNC_TO_LDS_B128
//     (ASYNCcnt / s_wait_asynccnt path)
//   - fused flash attention (S^T = K*Q^T so softmax is per-lane; O^T = V^T*P^T)
// ---------------------------------------------------------------------------

#define WIDTH_   1024
#define HEADS_   16
#define ATTN_CH  64
#define TSEQ     2048
#define NBATCH   2

typedef __attribute__((ext_vector_type(16))) __bf16 v16bf;
typedef __attribute__((ext_vector_type(8)))  float  v8f;

union Frag {
  v16bf v;
  uint4 q[2];
  unsigned int u[8];
  __bf16 h[16];
};

__device__ __forceinline__ v8f wmma_bf16(const Frag& a, const Frag& b, v8f c) {
  // D = A(16x32 bf16) x B(32x16 bf16) + C(16x16 f32)
  return __builtin_amdgcn_wmma_f32_16x16x32_bf16(
      /*neg_a=*/false, a.v, /*neg_b=*/false, b.v,
      /*c_mod=*/(short)0, c, /*reuse_a=*/false, /*reuse_b=*/false);
}

__device__ __forceinline__ unsigned pack_bf16(float lo, float hi) {
  union { __bf16 h2[2]; unsigned u; } p;
  p.h2[0] = (__bf16)lo; p.h2[1] = (__bf16)hi;
  return p.u;
}

// 16-byte async copy global -> LDS (per-lane). Low 32 bits of a generic LDS
// pointer are the DS byte address (aperture rule), which is what VDST wants.
__device__ __forceinline__ void async_load_b128(const void* g, void* l) {
  const unsigned long long ga = (unsigned long long)g;
  const unsigned lo = (unsigned)(size_t)l;
  asm volatile("global_load_async_to_lds_b128 %0, %1, off"
               :: "v"(lo), "v"(ga) : "memory");
}
__device__ __forceinline__ void wait_async() {
  asm volatile("s_wait_asynccnt 0" ::: "memory");
}

// ---------------------------------------------------------------------------
// Elementwise f32 -> bf16 (8 elements / thread)
// ---------------------------------------------------------------------------
__global__ __launch_bounds__(256) void cvt_bf16_kernel(
    const float* __restrict__ in, __bf16* __restrict__ out) {
  const long i = ((long)blockIdx.x * 256 + threadIdx.x) * 8;
  const float4 a = *(const float4*)(in + i);
  const float4 b = *(const float4*)(in + i + 4);
  union { __bf16 h8[8]; uint4 q; } p;
  p.h8[0] = (__bf16)a.x; p.h8[1] = (__bf16)a.y;
  p.h8[2] = (__bf16)a.z; p.h8[3] = (__bf16)a.w;
  p.h8[4] = (__bf16)b.x; p.h8[5] = (__bf16)b.y;
  p.h8[6] = (__bf16)b.z; p.h8[7] = (__bf16)b.w;
  *(uint4*)(out + i) = p.q;
}

// ---------------------------------------------------------------------------
// Transpose + convert: Wt[n][k] = (bf16) W[k][n].  32x32 LDS tiles.
// ---------------------------------------------------------------------------
__global__ __launch_bounds__(256) void transpose_cvt_kernel(
    const float* __restrict__ W, __bf16* __restrict__ Wt, int K, int N) {
  __shared__ float tile[32][33];
  const int bx = blockIdx.x * 32;            // n
  const int by = blockIdx.y * 32;            // k
  const int tx = threadIdx.x & 31;
  const int ty = threadIdx.x >> 5;           // 0..7
#pragma unroll
  for (int i = 0; i < 32; i += 8)
    tile[ty + i][tx] = W[(size_t)(by + ty + i) * N + bx + tx];
  __syncthreads();
#pragma unroll
  for (int i = 0; i < 32; i += 8)
    Wt[(size_t)(bx + ty + i) * K + by + tx] = (__bf16)tile[tx][ty + i];
}

// ---------------------------------------------------------------------------
// GEMM + bias:  Out[M,N] = X[M,1024](bf16) * Wt[N,1024](bf16)^T + b
// Block: 256 threads = 8 waves; tile 128(M) x 64(N), BK=32.
// A/B tiles staged with async-to-LDS 16B chunks; wave owns 2x2 WMMA tiles.
// ---------------------------------------------------------------------------
template <bool BF16OUT>
__global__ __launch_bounds__(256) void gemm_bias_kernel(
    const __bf16* __restrict__ X, const __bf16* __restrict__ Wt,
    const float* __restrict__ bias, void* __restrict__ Out, int N) {
  constexpr int K = 1024, LDA = 40, LDB = 40;
  __shared__ __align__(16) __bf16 As[128 * LDA];  // row-major, padded
  __shared__ __align__(16) __bf16 Bs[64 * LDB];   // [n][k], padded

  const int tid  = threadIdx.x;
  const int lane = tid & 31;
  const int wave = tid >> 5;
  const int wm   = (wave & 3) * 32;
  const int wn   = (wave >> 2) * 32;
  const int l15  = lane & 15;
  const int hi   = lane >> 4;
  const long blockM = (long)blockIdx.y * 128;
  const long blockN = (long)blockIdx.x * 64;
  const int o0 = hi ? 8 : 0;    // A-fragment K-chunk offsets (ISA layout)
  const int o1 = hi ? 24 : 16;

  // 16B-chunk staging assignments
  const int sr = tid >> 2;            // 0..63  (row / col index)
  const int sc = (tid & 3) * 8;       // 0,8,16,24 (k element offset)

  v8f acc[2][2] = {};

  for (int k0 = 0; k0 < K; k0 += 32) {
    // A tile: 128 rows x 32 k = 512 chunks (2/thread); B tile: 64 x 32 = 256.
    async_load_b128(X + (blockM + sr) * (long)K + k0 + sc, &As[sr * LDA + sc]);
    async_load_b128(X + (blockM + 64 + sr) * (long)K + k0 + sc,
                    &As[(64 + sr) * LDA + sc]);
    async_load_b128(Wt + (blockN + sr) * (long)K + k0 + sc, &Bs[sr * LDB + sc]);
    wait_async();
    __syncthreads();

    Frag af[2], bf[2];
#pragma unroll
    for (int i = 0; i < 2; ++i) {
      const __bf16* arow = &As[(wm + i * 16 + l15) * LDA];
      af[i].q[0] = *(const uint4*)(arow + o0);
      af[i].q[1] = *(const uint4*)(arow + o1);
    }
#pragma unroll
    for (int j = 0; j < 2; ++j) {
      const __bf16* brow = &Bs[(wn + j * 16 + l15) * LDB];
      bf[j].q[0] = *(const uint4*)(brow + o0);
      bf[j].q[1] = *(const uint4*)(brow + o1);
    }
#pragma unroll
    for (int i = 0; i < 2; ++i)
#pragma unroll
      for (int j = 0; j < 2; ++j) acc[i][j] = wmma_bf16(af[i], bf[j], acc[i][j]);
    __syncthreads();
  }

  // C/D layout: vgpr r -> row r (+8 for upper lane half), col = lane%16
#pragma unroll
  for (int i = 0; i < 2; ++i) {
    const long rbase = blockM + wm + i * 16 + hi * 8;
#pragma unroll
    for (int j = 0; j < 2; ++j) {
      const long col = blockN + wn + j * 16 + l15;
      const float bv = bias[col];
#pragma unroll
      for (int r = 0; r < 8; ++r) {
        const float val = acc[i][j][r] + bv;
        const long idx = (rbase + r) * (long)N + col;
        if (BF16OUT) ((__bf16*)Out)[idx] = (__bf16)val;
        else         ((float*)Out)[idx]  = val;
      }
    }
  }
}

// ---------------------------------------------------------------------------
// Flash attention per (b, h): S^T = K*Q^T (softmax rows become per-lane
// columns -> one shfl_xor(16) per reduction), online softmax, O^T += V^T*P^T
// with V^T staged transposed in LDS. 8 waves x 16 q rows per block.
// ---------------------------------------------------------------------------
__global__ __launch_bounds__(256) void attention_kernel(
    const __bf16* __restrict__ Q, const __bf16* __restrict__ KV,
    __bf16* __restrict__ O) {
  constexpr int LDV = 40;
  __shared__ __align__(16) __bf16 VT[ATTN_CH * LDV];

  const int tid = threadIdx.x;
  const int lane = tid & 31;
  const int wave = tid >> 5;
  const int l15 = lane & 15;
  const int hi = lane >> 4;
  const int b = blockIdx.z;
  const int h = blockIdx.y;
  const int qBase = blockIdx.x * 128 + wave * 16;

  const __bf16* qp = Q + ((long)(b * TSEQ + qBase)) * WIDTH_ + h * ATTN_CH;
  const __bf16* kp = KV + (long)b * TSEQ * (2 * WIDTH_) + h * ATTN_CH;
  const __bf16* vp = kp + WIDTH_;

  // Q^T as B-fragments, held across the whole Tk loop
  Frag qf[2];
  {
    const __bf16* qrow = qp + (long)l15 * WIDTH_ + hi * 8;
    qf[0].q[0] = *(const uint4*)(qrow);
    qf[0].q[1] = *(const uint4*)(qrow + 16);
    qf[1].q[0] = *(const uint4*)(qrow + 32);
    qf[1].q[1] = *(const uint4*)(qrow + 48);
  }

  const int vtk = tid & 31;
  const int vch = (tid >> 5) * 8;
  const int o0 = hi ? 8 : 0;
  const int o1 = hi ? 24 : 16;

  float mrow = -1e30f, lrow = 0.f;
  v8f oacc[4] = {};
  const float scale2 = 1.0f / 64.0f;  // reference scales q AND k by 1/8

  for (int t0 = 0; t0 < TSEQ; t0 += 32) {
    // stage V^T[ch][tk] (32 keys x 64 ch), transposing on store
    const __bf16* vr = vp + (long)(t0 + vtk) * (2 * WIDTH_) + vch;
#pragma unroll
    for (int i = 0; i < 8; ++i) VT[(vch + i) * LDV + vtk] = vr[i];
    __syncthreads();

    v8f s0 = {}, s1 = {};
    {
      const __bf16* k0r = kp + (long)(t0 + l15) * (2 * WIDTH_);
      const __bf16* k1r = k0r + 16 * (2 * WIDTH_);
      Frag kf;
      kf.q[0] = *(const uint4*)(k0r + o0);
      kf.q[1] = *(const uint4*)(k0r + o1);
      s0 = wmma_bf16(kf, qf[0], s0);
      kf.q[0] = *(const uint4*)(k0r + 32 + o0);
      kf.q[1] = *(const uint4*)(k0r + 32 + o1);
      s0 = wmma_bf16(kf, qf[1], s0);
      kf.q[0] = *(const uint4*)(k1r + o0);
      kf.q[1] = *(const uint4*)(k1r + o1);
      s1 = wmma_bf16(kf, qf[0], s1);
      kf.q[0] = *(const uint4*)(k1r + 32 + o0);
      kf.q[1] = *(const uint4*)(k1r + 32 + o1);
      s1 = wmma_bf16(kf, qf[1], s1);
    }

    // online softmax for this lane's tq column
    float tmax = -1e30f;
#pragma unroll
    for (int r = 0; r < 8; ++r) {
      s0[r] *= scale2; s1[r] *= scale2;
      tmax = fmaxf(tmax, fmaxf(s0[r], s1[r]));
    }
    tmax = fmaxf(tmax, __shfl_xor(tmax, 16, 32));
    const float mnew = fmaxf(mrow, tmax);
    const float corr = __expf(mrow - mnew);
    float p0[8], p1[8], psum = 0.f;
#pragma unroll
    for (int r = 0; r < 8; ++r) {
      p0[r] = __expf(s0[r] - mnew);
      p1[r] = __expf(s1[r] - mnew);
      psum += p0[r] + p1[r];
    }
    psum += __shfl_xor(psum, 16, 32);
    lrow = lrow * corr + psum;
    mrow = mnew;
#pragma unroll
    for (int c = 0; c < 4; ++c)
#pragma unroll
      for (int r = 0; r < 8; ++r) oacc[c][r] *= corr;

    // rebuild P^T (C-layout) as a B-fragment: swap row-halves across lanes
    unsigned c0[4], c1[4];
#pragma unroll
    for (int i = 0; i < 4; ++i) {
      c0[i] = pack_bf16(p0[2 * i], p0[2 * i + 1]);
      c1[i] = pack_bf16(p1[2 * i], p1[2 * i + 1]);
    }
    Frag pf;
#pragma unroll
    for (int i = 0; i < 4; ++i) {
      const unsigned x0 = (unsigned)__shfl_xor((int)c0[i], 16, 32);
      const unsigned x1 = (unsigned)__shfl_xor((int)c1[i], 16, 32);
      pf.u[i]     = hi ? x1    : c0[i];  // K 0..7  | 16..23
      pf.u[4 + i] = hi ? c1[i] : x0;     // K 8..15 | 24..31
    }

    // O^T[ch,tq] += V^T(16x32) * P^T(32x16), 4 ch-tiles
#pragma unroll
    for (int c = 0; c < 4; ++c) {
      const __bf16* vrow = &VT[(c * 16 + l15) * LDV];
      Frag vf;
      vf.q[0] = *(const uint4*)(vrow + hi * 16);
      vf.q[1] = *(const uint4*)(vrow + hi * 16 + 8);
      oacc[c] = wmma_bf16(vf, pf, oacc[c]);
    }
    __syncthreads();
  }

  const float inv = 1.0f / lrow;
  __bf16* op =
      O + ((long)(b * TSEQ + qBase + l15)) * WIDTH_ + h * ATTN_CH + hi * 8;
#pragma unroll
  for (int c = 0; c < 4; ++c) {
    union { __bf16 h8[8]; uint4 q; } pk;
#pragma unroll
    for (int r = 0; r < 8; ++r) pk.h8[r] = (__bf16)(oacc[c][r] * inv);
    *(uint4*)(op + c * 16) = pk.q;
  }
}

// ---------------------------------------------------------------------------
extern "C" void kernel_launch(void* const* d_in, const int* in_sizes, int n_in,
                              void* d_out, int out_size, void* d_ws,
                              size_t ws_size, hipStream_t stream) {
  (void)in_sizes; (void)n_in; (void)out_size; (void)ws_size;
  const float* x   = (const float*)d_in[0];
  const float* tgt = (const float*)d_in[1];
  const float* Wq  = (const float*)d_in[2];
  const float* bq  = (const float*)d_in[3];
  const float* Wkv = (const float*)d_in[4];
  const float* bkv = (const float*)d_in[5];
  const float* Wp  = (const float*)d_in[6];
  const float* bp  = (const float*)d_in[7];
  float* out = (float*)d_out;

  char* ws = (char*)d_ws;
  __bf16* xb   = (__bf16*)(ws);                        //  8 MiB [4096,1024]
  __bf16* tb   = (__bf16*)(ws + (size_t)( 8u << 20));  //  8 MiB [4096,1024]
  __bf16* Wqt  = (__bf16*)(ws + (size_t)(16u << 20));  //  2 MiB [1024,1024]
  __bf16* Wkvt = (__bf16*)(ws + (size_t)(18u << 20));  //  4 MiB [2048,1024]
  __bf16* Wpt  = (__bf16*)(ws + (size_t)(22u << 20));  //  2 MiB [1024,1024]
  __bf16* qb   = (__bf16*)(ws + (size_t)(24u << 20));  //  8 MiB [4096,1024]
  __bf16* kvb  = (__bf16*)(ws + (size_t)(32u << 20));  // 16 MiB [4096,2048]
  __bf16* ab   = (__bf16*)(ws + (size_t)(48u << 20));  //  8 MiB [4096,1024]

  const dim3 blk(256);
  const int M = NBATCH * TSEQ;  // 4096

  // one-time conversions (memory-bound, ~50 MB total at 23.3 TB/s)
  cvt_bf16_kernel<<<dim3((M * WIDTH_) / 2048), blk, 0, stream>>>(x, xb);
  cvt_bf16_kernel<<<dim3((M * WIDTH_) / 2048), blk, 0, stream>>>(tgt, tb);
  transpose_cvt_kernel<<<dim3(32, 32), blk, 0, stream>>>(Wq, Wqt, 1024, 1024);
  transpose_cvt_kernel<<<dim3(64, 32), blk, 0, stream>>>(Wkv, Wkvt, 1024, 2048);
  transpose_cvt_kernel<<<dim3(32, 32), blk, 0, stream>>>(Wp, Wpt, 1024, 1024);

  // Q = x*Wq + bq -> bf16
  gemm_bias_kernel<true>
      <<<dim3(WIDTH_ / 64, M / 128), blk, 0, stream>>>(xb, Wqt, bq, qb, WIDTH_);
  // KV = tgt*Wkv + bkv -> bf16 ([..,0:1024]=K, [..,1024:2048]=V)
  gemm_bias_kernel<true>
      <<<dim3(2 * WIDTH_ / 64, M / 128), blk, 0, stream>>>(tb, Wkvt, bkv, kvb,
                                                           2 * WIDTH_);
  // fused flash attention -> bf16
  attention_kernel<<<dim3(TSEQ / 128, HEADS_, NBATCH), blk, 0, stream>>>(qb, kvb,
                                                                         ab);
  // out = attn*Wp + bp -> f32
  gemm_bias_kernel<false>
      <<<dim3(WIDTH_ / 64, M / 128), blk, 0, stream>>>(ab, Wpt, bp, out, WIDTH_);
}
